// SAGE_76802605187215
// MI455X (gfx1250) — matrix operand order repaired and checked by reading, not compiled, not run on hardware
//
#include <hip/hip_runtime.h>
#include <hip/hip_bf16.h>
#include <stdint.h>

// ---------------------------------------------------------------------------
// GraphSAGE 2-layer forward for MI455X (gfx1250, wave32, WMMA).
// GEMMs use v_wmma_f32_16x16x32_bf16 with weights pre-swizzled into the
// B-fragment register layout (2x global_load_b128 per fragment per lane).
// Scatter/aggregation uses f32 atomics (node tensors fit in the 192MB L2).
// ---------------------------------------------------------------------------

typedef __bf16 bf16_t;
typedef __attribute__((ext_vector_type(16))) __bf16 v16bf;
typedef __attribute__((ext_vector_type(8)))  float  v8f;

#define NN      100000
#define NE      1600000
#define DIN     64
#define HID     256
#define EPSB    1e-5f
#define SLOPE   0.01f

// workspace layout in floats
#define OFF_DEG   0
#define OFF_AGG1  (OFF_DEG  + NN)            // N x 64
#define OFF_AGG2  (OFF_AGG1 + NN * DIN)      // N x 256
#define OFF_H     (OFF_AGG2 + NN * HID)      // N x 256 (layer-1 activations)
#define OFF_SUM1  (OFF_H    + NN * HID)      // 256
#define OFF_SQ1   (OFF_SUM1 + HID)           // 256
#define OFF_SUM2  (OFF_SQ1  + HID)           // 64
#define OFF_SQ2   (OFF_SUM2 + DIN)           // 64
#define OFF_END   (OFF_SQ2  + DIN)
// bf16 swizzled weights appended after OFF_END floats (each 64*256 elems):
#define WB_L1 0
#define WB_R1 (64 * 256)
#define WB_L2 (2 * 64 * 256)
#define WB_R2 (3 * 64 * 256)

// ---------------------------------------------------------------------------
// small helpers
// ---------------------------------------------------------------------------

__global__ void zero_ws_kernel(float* wsf, int count) {
    int stride = gridDim.x * blockDim.x;
    for (int i = blockIdx.x * blockDim.x + threadIdx.x; i < count; i += stride)
        wsf[i] = 0.0f;
}

// Convert f32 weight W[K][C] into bf16 fragments laid out exactly as the
// WMMA B-operand registers: fragment f = colTile*(K/32) + kChunk holds
// 32 lanes x 16 bf16; lane = (half,n): element 2v(+1) = W[kc*32+half*16+2v(+1)][ct*16+n].
// One thread per bf16 pair: p = (f*32 + lane)*8 + v.
__global__ void cvt_swz_kernel(const float* __restrict__ src,
                               bf16_t* __restrict__ dst, int K, int C) {
    int p = blockIdx.x * blockDim.x + threadIdx.x;
    int total = (K * C) >> 1;
    if (p < total) {
        int KC   = K >> 5;
        int v    = p & 7;
        int lane = (p >> 3) & 31;
        int f    = p >> 8;
        int kc   = f % KC;
        int ct   = f / KC;
        int half = lane >> 4;
        int n    = lane & 15;
        int c    = ct * 16 + n;
        int k    = kc * 32 + half * 16 + 2 * v;
        dst[2 * p]     = (__bf16)src[k * C + c];
        dst[2 * p + 1] = (__bf16)src[(k + 1) * C + c];
    }
}

__global__ void degree_kernel(const long long* __restrict__ ei,
                              float* __restrict__ deg) {
    int e = blockIdx.x * blockDim.x + threadIdx.x;
    if (e < NE) {
        int d = (int)ei[NE + e];
        atomicAdd(&deg[d], 1.0f);
    }
}

// scatter-add of 64-wide features: agg1[dst] += x[src]
__global__ void scatter64_kernel(const long long* __restrict__ ei,
                                 const float* __restrict__ x,
                                 float* __restrict__ agg) {
    int gid = blockIdx.x * blockDim.x + threadIdx.x;
    if (gid < NE * DIN) {
        int e = gid >> 6;
        int f = gid & 63;
        int s = (int)ei[e];
        int d = (int)ei[NE + e];
        atomicAdd(&agg[d * DIN + f], x[s * DIN + f]);
    }
}

// scatter-add of 256-wide features: agg2[dst] += h[src]
__global__ void scatter256_kernel(const long long* __restrict__ ei,
                                  const float* __restrict__ h,
                                  float* __restrict__ agg) {
    int gid = blockIdx.x * blockDim.x + threadIdx.x;
    if (gid < NE * HID) {
        int e = gid >> 8;
        int f = gid & 255;
        int s = (int)ei[e];
        int d = (int)ei[NE + e];
        const float* p = &h[s * HID + f];
        __builtin_prefetch(p, 0, 1);   // global_prefetch path
        atomicAdd(&agg[d * HID + f], *p);
    }
}

// ---------------------------------------------------------------------------
// WMMA fragment loaders
//   A 16x32: lane = (half,row); VGPR j: K = half*16 + 2j(+1) (matches B swizzle).
//   Loaded from f32, scaled, cvt bf16.
//   B: pre-swizzled -> one 32B vector load per lane (2x global_load_b128).
// ---------------------------------------------------------------------------

__device__ __forceinline__ v16bf load_a_frag(const float* __restrict__ rowp,
                                             int k0, int half, float scale) {
    v16bf a;
#pragma unroll
    for (int j = 0; j < 8; ++j) {
        int k = k0 + half * 16 + 2 * j;
        a[2 * j]     = (__bf16)(rowp[k] * scale);
        a[2 * j + 1] = (__bf16)(rowp[k + 1] * scale);
    }
    return a;
}

__device__ __forceinline__ v16bf load_b_frag_swz(const bf16_t* __restrict__ Wswz,
                                                 int fragIdx, int lane) {
    return *(const v16bf*)(Wswz + (size_t)(fragIdx * 32 + lane) * 16);
}

__device__ __forceinline__ v8f wmma_bf16(v16bf a, v16bf b, v8f c) {
    return __builtin_amdgcn_wmma_f32_16x16x32_bf16(false, a, false, b,
                                                   (short)0, c, false, false);
}

// ---------------------------------------------------------------------------
// Layer-1 GEMM: h_pre = (agg1/deg) @ Wl1 + bl1 + x @ Wr1   (N x 256, K = 64)
// wave = (rowTile, group of 4 colTiles): A fragments loaded once per K-chunk,
// reused across 4 accumulators -> 16 WMMAs per wave. Fused bias + BN stats.
// ---------------------------------------------------------------------------
__global__ void gemm1_kernel(const float* __restrict__ agg1,
                             const float* __restrict__ deg,
                             const float* __restrict__ x,
                             const bf16_t* __restrict__ Wl,
                             const bf16_t* __restrict__ Wr,
                             const float* __restrict__ bl,
                             float* __restrict__ hout,
                             float* __restrict__ sum, float* __restrict__ sq) {
    const int wave = (blockIdx.x * blockDim.x + threadIdx.x) >> 5;
    const int NW = (NN / 16) * 4;              // 4 col-groups of 4 tiles
    if (wave >= NW) return;                    // wave-uniform exit
    const int rowTile = wave >> 2;
    const int cg      = wave & 3;              // col group: tiles cg*4 .. cg*4+3

    const int lane = threadIdx.x & 31;
    const int n    = lane & 15;
    const int half = lane >> 4;
    const int row  = rowTile * 16 + n;

    const float rdeg = 1.0f / fmaxf(deg[row], 1.0f);
    const float* arow_m = agg1 + row * DIN;
    const float* arow_x = x    + row * DIN;

    v8f acc[4];
#pragma unroll
    for (int t = 0; t < 4; ++t) {
        const float bias = bl[(cg * 4 + t) * 16 + n];
#pragma unroll
        for (int v = 0; v < 8; ++v) acc[t][v] = bias;
    }

#pragma unroll
    for (int kc = 0; kc < 2; ++kc) {           // K = 64 -> 2 chunks
        v16bf aM = load_a_frag(arow_m, kc * 32, half, rdeg);
        v16bf aX = load_a_frag(arow_x, kc * 32, half, 1.0f);
#pragma unroll
        for (int t = 0; t < 4; ++t) {
            const int ct = cg * 4 + t;
            acc[t] = wmma_bf16(aM, load_b_frag_swz(Wl, ct * 2 + kc, lane), acc[t]);
            acc[t] = wmma_bf16(aX, load_b_frag_swz(Wr, ct * 2 + kc, lane), acc[t]);
        }
    }

#pragma unroll
    for (int t = 0; t < 4; ++t) {
        const int c = (cg * 4 + t) * 16 + n;
        float lsum = 0.0f, lsq = 0.0f;
#pragma unroll
        for (int v = 0; v < 8; ++v) {
            const int m = rowTile * 16 + v + 8 * half;
            hout[m * HID + c] = acc[t][v];
            lsum += acc[t][v];
            lsq  += acc[t][v] * acc[t][v];
        }
        lsum += __shfl_xor(lsum, 16, 32);      // combine lane halves
        lsq  += __shfl_xor(lsq, 16, 32);
        if (half == 0) {
            atomicAdd(&sum[c], lsum);
            atomicAdd(&sq[c],  lsq);
        }
    }
}

// ---------------------------------------------------------------------------
// Layer-2 GEMM: out_pre = (agg2/deg) @ Wl2 + bl2 + h @ Wr2  (N x 64, K = 256)
// wave = rowTile, all 4 colTiles resident -> 64 WMMAs per wave.
// ---------------------------------------------------------------------------
__global__ void gemm2_kernel(const float* __restrict__ agg2,
                             const float* __restrict__ deg,
                             const float* __restrict__ h,
                             const bf16_t* __restrict__ Wl,
                             const bf16_t* __restrict__ Wr,
                             const float* __restrict__ bl,
                             float* __restrict__ out,
                             float* __restrict__ sum, float* __restrict__ sq) {
    const int wave = (blockIdx.x * blockDim.x + threadIdx.x) >> 5;
    const int NW = NN / 16;                    // 6250 row tiles
    if (wave >= NW) return;                    // wave-uniform exit
    const int rowTile = wave;

    const int lane = threadIdx.x & 31;
    const int n    = lane & 15;
    const int half = lane >> 4;
    const int row  = rowTile * 16 + n;

    const float rdeg = 1.0f / fmaxf(deg[row], 1.0f);
    const float* arow_m = agg2 + row * HID;
    const float* arow_h = h    + row * HID;

    v8f acc[4];
#pragma unroll
    for (int t = 0; t < 4; ++t) {
        const float bias = bl[t * 16 + n];
#pragma unroll
        for (int v = 0; v < 8; ++v) acc[t][v] = bias;
    }

#pragma unroll
    for (int kc = 0; kc < 8; ++kc) {           // K = 256 -> 8 chunks
        v16bf aM = load_a_frag(arow_m, kc * 32, half, rdeg);
        v16bf aH = load_a_frag(arow_h, kc * 32, half, 1.0f);
#pragma unroll
        for (int t = 0; t < 4; ++t) {
            acc[t] = wmma_bf16(aM, load_b_frag_swz(Wl, t * 8 + kc, lane), acc[t]);
            acc[t] = wmma_bf16(aH, load_b_frag_swz(Wr, t * 8 + kc, lane), acc[t]);
        }
    }

#pragma unroll
    for (int t = 0; t < 4; ++t) {
        const int c = t * 16 + n;
        float lsum = 0.0f, lsq = 0.0f;
#pragma unroll
        for (int v = 0; v < 8; ++v) {
            const int m = rowTile * 16 + v + 8 * half;
            out[m * DIN + c] = acc[t][v];
            lsum += acc[t][v];
            lsq  += acc[t][v] * acc[t][v];
        }
        lsum += __shfl_xor(lsum, 16, 32);
        lsq  += __shfl_xor(lsq, 16, 32);
        if (half == 0) {
            atomicAdd(&sum[c], lsum);
            atomicAdd(&sq[c],  lsq);
        }
    }
}

// ---------------------------------------------------------------------------
// BN (from accumulated sum/sumsq) + leaky relu, in place. C is a power of 2.
// ---------------------------------------------------------------------------
__global__ void bn_lrelu_kernel(float* __restrict__ data,
                                const float* __restrict__ sum,
                                const float* __restrict__ sq,
                                const float* __restrict__ g,
                                const float* __restrict__ b,
                                int cmask, int total) {
    int i = blockIdx.x * blockDim.x + threadIdx.x;
    if (i < total) {
        int c = i & cmask;
        const float invN = 1.0f / (float)NN;
        float mu  = sum[c] * invN;
        float var = sq[c] * invN - mu * mu;
        float t = (data[i] - mu) * rsqrtf(var + EPSB) * g[c] + b[c];
        data[i] = (t > 0.0f) ? t : SLOPE * t;
    }
}

// ---------------------------------------------------------------------------
// launch
// ---------------------------------------------------------------------------
extern "C" void kernel_launch(void* const* d_in, const int* in_sizes, int n_in,
                              void* d_out, int out_size, void* d_ws, size_t ws_size,
                              hipStream_t stream) {
    (void)in_sizes; (void)n_in; (void)out_size; (void)ws_size;

    const float*     x   = (const float*)d_in[0];
    const long long* ei  = (const long long*)d_in[1];
    const float*     Wl1 = (const float*)d_in[2];
    const float*     bl1 = (const float*)d_in[3];
    const float*     Wr1 = (const float*)d_in[4];
    const float*     g1  = (const float*)d_in[5];
    const float*     b1  = (const float*)d_in[6];
    const float*     Wl2 = (const float*)d_in[7];
    const float*     bl2 = (const float*)d_in[8];
    const float*     Wr2 = (const float*)d_in[9];
    const float*     g2  = (const float*)d_in[10];
    const float*     b2  = (const float*)d_in[11];

    float* wsf   = (float*)d_ws;
    float* deg   = wsf + OFF_DEG;
    float* agg1  = wsf + OFF_AGG1;
    float* agg2  = wsf + OFF_AGG2;
    float* h     = wsf + OFF_H;
    float* sum1  = wsf + OFF_SUM1;
    float* sq1   = wsf + OFF_SQ1;
    float* sum2  = wsf + OFF_SUM2;
    float* sq2   = wsf + OFF_SQ2;
    bf16_t* wb   = (bf16_t*)(wsf + OFF_END);   // 32B-aligned (OFF_END*4 % 32 == 0)
    bf16_t* Wl1b = wb + WB_L1;
    bf16_t* Wr1b = wb + WB_R1;
    bf16_t* Wl2b = wb + WB_L2;
    bf16_t* Wr2b = wb + WB_R2;
    float* out   = (float*)d_out;

    const int TPB = 256;

    // 0) zero accumulators + stats (every call: deterministic)
    zero_ws_kernel<<<2048, TPB, 0, stream>>>(wsf, OFF_END);

    // 1) weights -> bf16, pre-swizzled into B-fragment layout
    const int WPAIRS = (DIN * HID) / 2;  // 8192 pairs each
    cvt_swz_kernel<<<(WPAIRS + TPB - 1) / TPB, TPB, 0, stream>>>(Wl1, Wl1b, DIN, HID);
    cvt_swz_kernel<<<(WPAIRS + TPB - 1) / TPB, TPB, 0, stream>>>(Wr1, Wr1b, DIN, HID);
    cvt_swz_kernel<<<(WPAIRS + TPB - 1) / TPB, TPB, 0, stream>>>(Wl2, Wl2b, HID, DIN);
    cvt_swz_kernel<<<(WPAIRS + TPB - 1) / TPB, TPB, 0, stream>>>(Wr2, Wr2b, HID, DIN);

    // 2) degrees + layer-1 scatter
    degree_kernel<<<(NE + TPB - 1) / TPB, TPB, 0, stream>>>(ei, deg);
    scatter64_kernel<<<(NE * DIN) / TPB, TPB, 0, stream>>>(ei, x, agg1);

    // 3) layer-1 GEMM (WMMA) with fused bias + BN stats
    {
        const int waves = (NN / 16) * 4;             // 25000, /8 = 3125 blocks
        gemm1_kernel<<<waves / 8, TPB, 0, stream>>>(agg1, deg, x, Wl1b, Wr1b,
                                                    bl1, h, sum1, sq1);
    }

    // 4) BN1 + leaky relu in place on h
    bn_lrelu_kernel<<<(NN * HID) / TPB, TPB, 0, stream>>>(h, sum1, sq1, g1, b1,
                                                          HID - 1, NN * HID);

    // 5) layer-2 scatter (E x 256)
    scatter256_kernel<<<(NE * HID) / TPB, TPB, 0, stream>>>(ei, h, agg2);

    // 6) layer-2 GEMM (WMMA)
    {
        const int waves = NN / 16;                   // 6250 -> 782 blocks
        gemm2_kernel<<<(waves + 7) / 8, TPB, 0, stream>>>(agg2, deg, h, Wl2b, Wr2b,
                                                          bl2, out, sum2, sq2);
    }

    // 7) BN2 + leaky relu into d_out (in place)
    bn_lrelu_kernel<<<(NN * DIN) / TPB, TPB, 0, stream>>>(out, sum2, sq2, g2, b2,
                                                          DIN - 1, NN * DIN);
}
